// TimeDistributedCNN_32444182954216
// MI455X (gfx1250) — matrix operand order repaired and checked by reading, not compiled, not run on hardware
//
#include <hip/hip_runtime.h>

// ---------------- constants ----------------
#define B_   8
#define T_   2000
#define W_   50
#define EPSV 1e-5f

// workspace layout (bytes)
#define W2R_OFF   0        // bf16 [7][32][64]  = 28672 B
#define W3R_OFF   28672    // bf16 [5][16][32]  = 5120 B
#define SC_OFF    33792    // f32  s1[64] sh1[64] s2[32] sh2[32] s3[16] sh3[16] = 896 B
#define FEATS_OFF 34688    // bf16 [16000][16]  = 512000 B

typedef __bf16 bf16;
typedef bf16         v16bf __attribute__((ext_vector_type(16)));
typedef float        v8f   __attribute__((ext_vector_type(8)));
typedef unsigned int v4u   __attribute__((ext_vector_type(4)));

// CDNA5 async global->LDS copy path (ASYNCcnt) via inline asm; device pass only.
#if defined(__gfx1250__) && defined(__has_builtin)
#if __has_builtin(__builtin_amdgcn_s_wait_asynccnt)
#define USE_ASYNC_LDS 1
#endif
#endif
#ifndef USE_ASYNC_LDS
#define USE_ASYNC_LDS 0
#endif

#if USE_ASYNC_LDS
// Copy 16B global -> LDS asynchronously (tracked by ASYNCcnt).
// LDS operand is the low 32 bits of the generic pointer (== flat->local addrspacecast).
static __device__ __forceinline__ void async_copy_b128(const void* gsrc, void* ldst) {
  unsigned lds_addr = (unsigned)(unsigned long long)(uintptr_t)ldst;
  unsigned long long gaddr = (unsigned long long)(uintptr_t)gsrc;
  asm volatile("global_load_async_to_lds_b128 %0, %1, off"
               :
               : "v"(lds_addr), "v"(gaddr)
               : "memory");
}
#endif

union BFrag { v16bf v; v4u q[2]; };

// load 16 bf16 as two 16B chunks (p0: halves [0..7], p1: halves [8..15])
static __device__ __forceinline__ v16bf lds_frag2(const bf16* p0, const bf16* p1) {
  BFrag f;
  f.q[0] = *(const v4u*)p0;
  f.q[1] = *(const v4u*)p1;
  return f.v;
}
static __device__ __forceinline__ v16bf zero_frag() {
  BFrag f; v4u z = {0u,0u,0u,0u};
  f.q[0] = z; f.q[1] = z;
  return f.v;
}
static __device__ __forceinline__ float sigm(float x) { return 1.f / (1.f + __expf(-x)); }

// ---------------- prep: fold BN, reorder + bf16-convert conv weights ----------------
__global__ void prep_kernel(const float* __restrict__ c2w, const float* __restrict__ c3w,
                            const float* __restrict__ c1b, const float* __restrict__ g1,
                            const float* __restrict__ be1, const float* __restrict__ m1,
                            const float* __restrict__ v1,
                            const float* __restrict__ c2b, const float* __restrict__ g2,
                            const float* __restrict__ be2, const float* __restrict__ m2,
                            const float* __restrict__ v2,
                            const float* __restrict__ c3b, const float* __restrict__ g3,
                            const float* __restrict__ be3, const float* __restrict__ m3,
                            const float* __restrict__ v3,
                            unsigned char* __restrict__ ws) {
  bf16*  w2r = (bf16*)(ws + W2R_OFF);
  bf16*  w3r = (bf16*)(ws + W3R_OFF);
  float* sc  = (float*)(ws + SC_OFF);
  int tid  = blockIdx.x * blockDim.x + threadIdx.x;
  int nthr = gridDim.x * blockDim.x;
  for (int j = tid; j < 7 * 32 * 64; j += nthr) {       // w2r[k][co][ci]
    int k = j >> 11, rem = j & 2047, co = rem >> 6, ci = rem & 63;
    w2r[j] = (bf16)c2w[co * 448 + ci * 7 + k];
  }
  for (int j = tid; j < 5 * 16 * 32; j += nthr) {       // w3r[k][co][ci]
    int k = j >> 9, rem = j & 511, co = rem >> 5, ci = rem & 31;
    w3r[j] = (bf16)c3w[co * 160 + ci * 5 + k];
  }
  for (int j = tid; j < 224; j += nthr) {
    float val;
    if (j < 64)        { int c = j;       val = g1[c] * rsqrtf(v1[c] + EPSV); }
    else if (j < 128)  { int c = j - 64;  float s = g1[c] * rsqrtf(v1[c] + EPSV); val = (c1b[c] - m1[c]) * s + be1[c]; }
    else if (j < 160)  { int c = j - 128; val = g2[c] * rsqrtf(v2[c] + EPSV); }
    else if (j < 192)  { int c = j - 160; float s = g2[c] * rsqrtf(v2[c] + EPSV); val = (c2b[c] - m2[c]) * s + be2[c]; }
    else if (j < 208)  { int c = j - 192; val = g3[c] * rsqrtf(v3[c] + EPSV); }
    else               { int c = j - 208; float s = g3[c] * rsqrtf(v3[c] + EPSV); val = (c3b[c] - m3[c]) * s + be3[c]; }
    sc[j] = val;
  }
}

// ---------------- per-window CNN feature extractor (WMMA bf16) ----------------
// 1 block = 1 (b,t) sample; 128 threads = 4 waves; wave w owns output-position N-tile w.
__global__ __launch_bounds__(128) void conv_kernel(const float* __restrict__ x,
                                                   const float* __restrict__ c1w,
                                                   unsigned char* __restrict__ ws) {
  __shared__ __align__(16) float win[64];
  __shared__ __align__(16) bf16  w2s[7 * 32 * 64];
  __shared__ __align__(16) bf16  w3s[5 * 16 * 32];
  __shared__ __align__(16) bf16  h1t[50 * 64];   // [l][ci] conv1 output (bf16)
  __shared__ __align__(16) bf16  h2t[50 * 32];   // [l][ci] conv2 output (bf16)
  __shared__ float feats[16];

  const bf16*  w2g    = (const bf16*)(ws + W2R_OFF);
  const bf16*  w3g    = (const bf16*)(ws + W3R_OFF);
  const float* sc     = (const float*)(ws + SC_OFF);
  bf16*        featsG = (bf16*)(ws + FEATS_OFF);

  const int s = blockIdx.x;
  const int b = s / T_, t = s % T_;
  const int tid = threadIdx.x;

  // stage weights to LDS; async path overlaps the copy with conv1 below
#if USE_ASYNC_LDS
  {
    const char* src2 = (const char*)w2g;
    char*       dst2 = (char*)w2s;
    for (int j = tid; j < (7 * 32 * 64) / 8; j += 128)
      async_copy_b128(src2 + j * 16, dst2 + j * 16);
    const char* src3 = (const char*)w3g;
    char*       dst3 = (char*)w3s;
    for (int j = tid; j < (5 * 16 * 32) / 8; j += 128)
      async_copy_b128(src3 + j * 16, dst3 + j * 16);
  }
#else
  {
    const v4u* src2 = (const v4u*)w2g; v4u* dst2 = (v4u*)w2s;
    for (int j = tid; j < (7 * 32 * 64) / 8; j += 128) dst2[j] = src2[j];
    const v4u* src3 = (const v4u*)w3g; v4u* dst3 = (v4u*)w3s;
    for (int j = tid; j < (5 * 16 * 32) / 8; j += 128) dst3[j] = src3[j];
  }
#endif
  if (tid < 64) {
    int gt = t - (W_ - 1) + tid;                 // left zero-padded window
    win[tid] = (tid < W_ && gt >= 0) ? x[b * T_ + gt] : 0.f;
  }
  if (tid < 16) feats[tid] = 0.f;
  __syncthreads();                               // win/feats visible (weights may still be in flight)

  const float *s1 = sc, *sh1 = sc + 64, *s2 = sc + 128, *sh2 = sc + 160,
              *s3 = sc + 192, *sh3 = sc + 208;

  // conv1 (Cin=1, K=9): VALU, write transposed bf16 h1t[l][co] — overlaps async weight DMA
  for (int idx = tid; idx < 50 * 64; idx += 128) {
    int l = idx >> 6, co = idx & 63;
    float acc = 0.f;
#pragma unroll
    for (int k = 0; k < 9; ++k) {
      int p = l + k - 4;
      if (p >= 0 && p < W_) acc += c1w[co * 9 + k] * win[p];
    }
    h1t[l * 64 + co] = (bf16)fmaxf(acc * s1[co] + sh1[co], 0.f);
  }
#if USE_ASYNC_LDS
  __builtin_amdgcn_s_wait_asynccnt(0);           // this wave's weight DMA landed
#endif
  __syncthreads();                               // h1t + all waves' weights ready

  const int lane = tid & 31;
  const int nt   = tid >> 5;                     // wave = output N-tile
  const int half = lane >> 4;                    // lane group
  const int col  = lane & 15;
  const int lrow = nt * 16 + col;                // output position l (0..63; >=50 masked)

  // conv2: y2[32x50] = sum_k W2k[32x64] @ shift_k(h1)[64x50], K split into 2 tiles of 32
  v8f acc0 = {0,0,0,0,0,0,0,0};
  v8f acc1 = {0,0,0,0,0,0,0,0};
#pragma unroll
  for (int k = 0; k < 7; ++k) {
    int row = lrow + k - 3;
    bool ok = (row >= 0) && (row < W_);
#pragma unroll
    for (int kt = 0; kt < 2; ++kt) {
      v16bf bfr;
      if (ok) {
        const bf16* bp = &h1t[row * 64 + kt * 32 + half * 8]; // K runs at +0,+16
        bfr = lds_frag2(bp, bp + 16);
      } else bfr = zero_frag();
      const bf16* ap0 = &w2s[k * 2048 + col * 64 + kt * 32 + half * 8]; // K runs at +0,+16
      v16bf a0 = lds_frag2(ap0, ap0 + 16);
      const bf16* ap1 = ap0 + 16 * 64;                                  // co += 16
      v16bf a1 = lds_frag2(ap1, ap1 + 16);
      acc0 = __builtin_amdgcn_wmma_f32_16x16x32_bf16(false, a0, false, bfr, (short)0, acc0, false, false);
      acc1 = __builtin_amdgcn_wmma_f32_16x16x32_bf16(false, a1, false, bfr, (short)0, acc1, false, false);
    }
  }
  if (lrow < W_) {
#pragma unroll
    for (int r = 0; r < 8; ++r) {
      int co = half * 8 + r;                                  // C layout: M = half*8 + r
      h2t[lrow * 32 + co]      = (bf16)fmaxf(acc0[r] * s2[co]      + sh2[co],      0.f);
      h2t[lrow * 32 + co + 16] = (bf16)fmaxf(acc1[r] * s2[co + 16] + sh2[co + 16], 0.f);
    }
  }
  __syncthreads();

  // conv3: y3[16x50] = sum_k W3k[16x32] @ shift_k(h2)[32x50]
  v8f acc3 = {0,0,0,0,0,0,0,0};
#pragma unroll
  for (int k = 0; k < 5; ++k) {
    int row = lrow + k - 2;
    bool ok = (row >= 0) && (row < W_);
    v16bf bfr;
    if (ok) {
      const bf16* bp = &h2t[row * 32 + half * 8];
      bfr = lds_frag2(bp, bp + 16);
    } else bfr = zero_frag();
    const bf16* ap = &w3s[k * 512 + col * 32 + half * 8];
    v16bf a = lds_frag2(ap, ap + 16);
    acc3 = __builtin_amdgcn_wmma_f32_16x16x32_bf16(false, a, false, bfr, (short)0, acc3, false, false);
  }
  if (lrow < W_) {
#pragma unroll
    for (int r = 0; r < 8; ++r) {
      int co = half * 8 + r;
      atomicAdd(&feats[co], fmaxf(acc3[r] * s3[co] + sh3[co], 0.f));   // mean-pool accumulate
    }
  }
  __syncthreads();
  if (tid < 16) featsG[s * 16 + tid] = (bf16)(feats[tid] * (1.f / 50.f));
}

// ---------------- 2-layer LSTM + head: 1 persistent WG, 8 waves, weights in VGPR fragments ----------------
__global__ __launch_bounds__(256) void lstm_kernel(const float* __restrict__ wih0, const float* __restrict__ whh0,
                                                   const float* __restrict__ bih0, const float* __restrict__ bhh0,
                                                   const float* __restrict__ wih1, const float* __restrict__ whh1,
                                                   const float* __restrict__ bih1, const float* __restrict__ bhh1,
                                                   const float* __restrict__ fc1w, const float* __restrict__ fc1b,
                                                   const float* __restrict__ fc2w, const float* __restrict__ fc2b,
                                                   const unsigned char* __restrict__ ws, float* __restrict__ out) {
  __shared__ __align__(16) bf16 X0[16 * 96];    // [batch][ x(16) | h0(64) | zero-pad(16) ]
  __shared__ __align__(16) bf16 X1[16 * 128];   // [batch][ h1out(64) | h1state(64) ]
  __shared__ float gates[8 * 256];
  __shared__ float cA[8 * 64], cB[8 * 64], hBf[8 * 64];
  __shared__ float bias0[256], bias1[256];
  __shared__ float fc1wS[32 * 64], fc1bS[32], fc2wS[64], fc2bS[2];
  __shared__ float hid[8 * 32];

  const bf16* featsG = (const bf16*)(ws + FEATS_OFF);
  const int tid  = threadIdx.x;
  const int lane = tid & 31;
  const int wv   = tid >> 5;       // 0..7; owns gate N-tiles wv and wv+8
  const int half = lane >> 4;
  const int col  = lane & 15;      // batch row for A, gate column for B/C

  for (int j = tid; j < 16 * 96;  j += 256) X0[j] = (bf16)0.f;
  for (int j = tid; j < 16 * 128; j += 256) X1[j] = (bf16)0.f;
  for (int j = tid; j < 512; j += 256) { cA[j] = 0.f; cB[j] = 0.f; hBf[j] = 0.f; }
  bias0[tid] = bih0[tid] + bhh0[tid];
  bias1[tid] = bih1[tid] + bhh1[tid];
  for (int j = tid; j < 2048; j += 256) fc1wS[j] = fc1w[j];
  if (tid < 32) fc1bS[tid] = fc1b[tid];
  if (tid < 64) fc2wS[tid] = fc2w[tid];
  if (tid < 2)  fc2bS[tid] = fc2b[tid];

  // preload step-invariant weight B-fragments (K-major concat [w_ih | w_hh]^T) into VGPRs
  v16bf bw0[2][3], bw1[2][4];
#pragma unroll
  for (int ni = 0; ni < 2; ++ni) {
    int n = (wv + ni * 8) * 16 + col;
#pragma unroll
    for (int kt = 0; kt < 3; ++kt) {
      v16bf f;
#pragma unroll
      for (int e = 0; e < 16; ++e) {
        int kk = kt * 32 + half * 8 + (e & 7) + (e >> 3) * 16;
        float v = 0.f;
        if (kk < 16)      v = wih0[n * 16 + kk];
        else if (kk < 80) v = whh0[n * 64 + (kk - 16)];
        f[e] = (bf16)v;
      }
      bw0[ni][kt] = f;
    }
#pragma unroll
    for (int kt = 0; kt < 4; ++kt) {
      v16bf f;
#pragma unroll
      for (int e = 0; e < 16; ++e) {
        int kk = kt * 32 + half * 8 + (e & 7) + (e >> 3) * 16;
        float v = (kk < 64) ? wih1[n * 64 + kk] : whh1[n * 64 + (kk - 64)];
        f[e] = (bf16)v;
      }
      bw1[ni][kt] = f;
    }
  }
  __syncthreads();

  for (int t = 0; t < T_; ++t) {
    if (tid < 128) {                                   // x-part of X0 <- features
      int b = tid >> 4, j = tid & 15;
      X0[b * 96 + j] = featsG[(b * T_ + t) * 16 + j];
    }
    __syncthreads();

    // ---- layer 0 GEMM: gates = X0[16x96] * W0^T[96x256]
    {
      v16bf a[3];
#pragma unroll
      for (int kt = 0; kt < 3; ++kt) {
        const bf16* ap = &X0[col * 96 + kt * 32 + half * 8];
        a[kt] = lds_frag2(ap, ap + 16);
      }
#pragma unroll
      for (int ni = 0; ni < 2; ++ni) {
        v8f acc = {0,0,0,0,0,0,0,0};
#pragma unroll
        for (int kt = 0; kt < 3; ++kt)
          acc = __builtin_amdgcn_wmma_f32_16x16x32_bf16(false, a[kt], false, bw0[ni][kt], (short)0, acc, false, false);
        if (half == 0) {
          int n = (wv + ni * 8) * 16 + col;
#pragma unroll
          for (int r = 0; r < 8; ++r) gates[r * 256 + n] = acc[r] + bias0[n];
        }
      }
    }
    __syncthreads();

    for (int j = tid; j < 512; j += 256) {             // layer-0 state update
      int b = j >> 6, jj = j & 63;
      float iv = sigm(gates[b * 256 + jj]);
      float fv = sigm(gates[b * 256 + 64 + jj]);
      float gv = tanhf(gates[b * 256 + 128 + jj]);
      float ov = sigm(gates[b * 256 + 192 + jj]);
      float c  = fv * cA[j] + iv * gv;
      cA[j] = c;
      bf16 h = (bf16)(ov * tanhf(c));
      X0[b * 96 + 16 + jj] = h;                        // recurrent input, next step
      X1[b * 128 + jj]     = h;                        // layer-1 input, this step
    }
    __syncthreads();

    // ---- layer 1 GEMM: gates = X1[16x128] * W1^T[128x256]
    {
      v16bf a[4];
#pragma unroll
      for (int kt = 0; kt < 4; ++kt) {
        const bf16* ap = &X1[col * 128 + kt * 32 + half * 8];
        a[kt] = lds_frag2(ap, ap + 16);
      }
#pragma unroll
      for (int ni = 0; ni < 2; ++ni) {
        v8f acc = {0,0,0,0,0,0,0,0};
#pragma unroll
        for (int kt = 0; kt < 4; ++kt)
          acc = __builtin_amdgcn_wmma_f32_16x16x32_bf16(false, a[kt], false, bw1[ni][kt], (short)0, acc, false, false);
        if (half == 0) {
          int n = (wv + ni * 8) * 16 + col;
#pragma unroll
          for (int r = 0; r < 8; ++r) gates[r * 256 + n] = acc[r] + bias1[n];
        }
      }
    }
    __syncthreads();

    for (int j = tid; j < 512; j += 256) {             // layer-1 state update
      int b = j >> 6, jj = j & 63;
      float iv = sigm(gates[b * 256 + jj]);
      float fv = sigm(gates[b * 256 + 64 + jj]);
      float gv = tanhf(gates[b * 256 + 128 + jj]);
      float ov = sigm(gates[b * 256 + 192 + jj]);
      float c  = fv * cB[j] + iv * gv;
      cB[j] = c;
      float h = ov * tanhf(c);
      hBf[j] = h;
      X1[b * 128 + 64 + jj] = (bf16)h;
    }
    __syncthreads();

    // ---- classifier head
    {
      int b = tid >> 5, jj = tid & 31;
      float acc = fc1bS[jj];
#pragma unroll 8
      for (int k = 0; k < 64; ++k) acc += fc1wS[jj * 64 + k] * hBf[b * 64 + k];
      hid[b * 32 + jj] = fmaxf(acc, 0.f);
    }
    __syncthreads();
    if (tid < 16) {
      int b = tid >> 1, cls = tid & 1;
      float acc = fc2bS[cls];
#pragma unroll
      for (int k = 0; k < 32; ++k) acc += fc2wS[cls * 32 + k] * hid[b * 32 + k];
      out[(b * T_ + t) * 2 + cls] = acc;
    }
    __syncthreads();
  }
}

// ---------------- launcher ----------------
extern "C" void kernel_launch(void* const* d_in, const int* in_sizes, int n_in,
                              void* d_out, int out_size, void* d_ws, size_t ws_size,
                              hipStream_t stream) {
  (void)in_sizes; (void)n_in; (void)out_size; (void)ws_size;
  const float* x    = (const float*)d_in[0];
  const float* c1w  = (const float*)d_in[1];
  const float* c1b  = (const float*)d_in[2];
  const float* g1   = (const float*)d_in[3];
  const float* be1  = (const float*)d_in[4];
  const float* m1   = (const float*)d_in[5];
  const float* v1   = (const float*)d_in[6];
  const float* c2w  = (const float*)d_in[7];
  const float* c2b  = (const float*)d_in[8];
  const float* g2   = (const float*)d_in[9];
  const float* be2  = (const float*)d_in[10];
  const float* m2   = (const float*)d_in[11];
  const float* v2   = (const float*)d_in[12];
  const float* c3w  = (const float*)d_in[13];
  const float* c3b  = (const float*)d_in[14];
  const float* g3   = (const float*)d_in[15];
  const float* be3  = (const float*)d_in[16];
  const float* m3   = (const float*)d_in[17];
  const float* v3   = (const float*)d_in[18];
  const float* wih0 = (const float*)d_in[19];
  const float* whh0 = (const float*)d_in[20];
  const float* bih0 = (const float*)d_in[21];
  const float* bhh0 = (const float*)d_in[22];
  const float* wih1 = (const float*)d_in[23];
  const float* whh1 = (const float*)d_in[24];
  const float* bih1 = (const float*)d_in[25];
  const float* bhh1 = (const float*)d_in[26];
  const float* fc1w = (const float*)d_in[27];
  const float* fc1b = (const float*)d_in[28];
  const float* fc2w = (const float*)d_in[29];
  const float* fc2b = (const float*)d_in[30];

  unsigned char* ws = (unsigned char*)d_ws;
  float* out = (float*)d_out;

  prep_kernel<<<32, 256, 0, stream>>>(c2w, c3w, c1b, g1, be1, m1, v1,
                                      c2b, g2, be2, m2, v2,
                                      c3b, g3, be3, m3, v3, ws);
  conv_kernel<<<B_ * T_, 128, 0, stream>>>(x, c1w, ws);
  lstm_kernel<<<1, 256, 0, stream>>>(wih0, whh0, bih0, bhh0,
                                     wih1, whh1, bih1, bhh1,
                                     fc1w, fc1b, fc2w, fc2b, ws, out);
}
